// RankGNN_26044681683633
// MI455X (gfx1250) — matrix operand order repaired and checked by reading, not compiled, not run on hardware
//
#include <hip/hip_runtime.h>
#include <hip/hip_bf16.h>
#include <math.h>

typedef __attribute__((ext_vector_type(16))) _Float16 v16h;
typedef __attribute__((ext_vector_type(8)))  float    v8f;

#define FIN 9

// ---------------- utility kernels ----------------

__global__ void k_zero(float* p, long n) {
    long i = (long)blockIdx.x * blockDim.x + threadIdx.x;
    long st = (long)gridDim.x * blockDim.x;
    for (; i < n; i += st) p[i] = 0.0f;
}

__global__ void k_deg(const int* __restrict__ dst, float* __restrict__ deg, int E) {
    int i = blockIdx.x * blockDim.x + threadIdx.x;
    int st = gridDim.x * blockDim.x;
    for (; i < E; i += st) atomicAdd(&deg[dst[i]], 1.0f);
}

__global__ void k_dinv(float* deg, int N) {
    int i = blockIdx.x * blockDim.x + threadIdx.x;
    if (i < N) deg[i] = rsqrtf(deg[i] + 1.0f);
}

// pad x (N x 9 f32) -> xpad (N x 32 f16, cols >= 9 zero)
__global__ void k_padx(const float* __restrict__ x, _Float16* __restrict__ xpad, int N) {
    int idx = blockIdx.x * blockDim.x + threadIdx.x;
    if (idx >= N * 32) return;
    int i = idx >> 5, c = idx & 31;
    xpad[idx] = (c < FIN) ? (_Float16)x[i * FIN + c] : (_Float16)0.0f;
}

// Pre-shuffle weights (K x ncol f32, row-major) into WMMA B-fragment order (f16).
// Bp[((kt*nct + ct)*32 + lane)*16 + j] = W[(kt*32 + (lane>>4)*16 + j)*ncol + ct*16 + (lane&15)]
__global__ void k_packB(const float* __restrict__ W, _Float16* __restrict__ Bp,
                        int ktiles, int nct, int ncol, int krows) {
    int total = ktiles * nct * 32 * 16;
    int tid = blockIdx.x * blockDim.x + threadIdx.x;
    if (tid >= total) return;
    int j    = tid & 15;
    int lane = (tid >> 4) & 31;
    int t    = tid >> 9;
    int ct   = t % nct;
    int kt   = t / nct;
    int k    = kt * 32 + (lane >> 4) * 16 + j;
    int col  = ct * 16 + (lane & 15);
    float v  = (k < krows) ? W[k * ncol + col] : 0.0f;
    Bp[tid]  = (_Float16)v;
}

// ---------------- WMMA GEMM ----------------
// A: nrows x (KTILES*32) f16 row-major. Bp: pre-shuffled wave32 B fragments.
// 64 rows per block (4 row-tiles), one wave per 16-wide column tile.
// Each B fragment is loaded once per k-step and reused across 4 row-tiles.
// FUSE=false: Cf32 = acc (raw pre-activation, f32)
// FUSE=true : Ch16 = tanh(acc + bias) (fused MLP epilogue, f16)
template<int KTILES, int NCTILES, bool FUSE>
__global__ void k_gemm_wmma(const _Float16* __restrict__ A,
                            const _Float16* __restrict__ Bp,
                            float* __restrict__ Cf,
                            _Float16* __restrict__ Ch,
                            const float* __restrict__ bias,
                            int nrows) {
    constexpr int K    = KTILES * 32;
    constexpr int ncol = NCTILES * 16;
    constexpr int ROWS = 64;
    __shared__ _Float16 sA[ROWS * K];   // <= 16 KB

    const int tid  = threadIdx.x;
    const int lane = tid & 31;
    const int wv   = tid >> 5;                    // column tile index
    const int rowBase = blockIdx.x * ROWS;
    const int valid   = nrows - rowBase;          // rows available in this block
    const _Float16* Ablk = A + (size_t)rowBase * K;

    // stage 64 x K A-tile into LDS (rows are contiguous at stride K)
    if (valid >= ROWS) {
        const uint4* s4 = reinterpret_cast<const uint4*>(Ablk);
        uint4*       d4 = reinterpret_cast<uint4*>(sA);
        for (int i = tid; i < ROWS * K / 8; i += (int)blockDim.x) d4[i] = s4[i];
    } else {
        const int validHalf = valid * K;
        for (int i = tid * 8; i < ROWS * K; i += (int)blockDim.x * 8) {
#pragma unroll
            for (int j = 0; j < 8; ++j)
                sA[i + j] = (i + j < validHalf) ? Ablk[i + j] : (_Float16)0.0f;
        }
    }
    __syncthreads();

    const int arow = lane & 15;          // row within 16-row tile (ISA A layout)
    const int kh   = (lane >> 4) * 16;   // K sub-offset (contiguous, matches packed B)
    v8f c[4] = {v8f{}, v8f{}, v8f{}, v8f{}};

#pragma unroll
    for (int kt = 0; kt < KTILES; ++kt) {
        const int k0 = kt * 32;
        v16h b = *reinterpret_cast<const v16h*>(Bp + ((kt * NCTILES + wv) * 32 + lane) * 16);
#pragma unroll
        for (int rt = 0; rt < 4; ++rt) {
            const int rbase = (rt * 16 + arow) * K + k0 + kh;
            v16h a;
#pragma unroll
            for (int j = 0; j < 8; ++j) {
                a[j]     = sA[rbase + j];        // K = kh+0..7   (VGPR0-3)
                a[8 + j] = sA[rbase + 8 + j];    // K = kh+8..15  (VGPR4-7)
            }
            c[rt] = __builtin_amdgcn_wmma_f32_16x16x32_f16(
                false, a, false, b, (short)0, c[rt], false, false);
        }
    }

    const int nloc  = wv * 16 + (lane & 15);
    const int mbase = (lane >> 4) * 8;
    const float bv  = FUSE ? bias[nloc] : 0.0f;

    if (valid >= ROWS) {                 // fast path: block fully in-bounds
#pragma unroll
        for (int rt = 0; rt < 4; ++rt) {
            const int r0 = rowBase + rt * 16 + mbase;
#pragma unroll
            for (int r = 0; r < 8; ++r) {
                if (FUSE) Ch[(r0 + r) * ncol + nloc] = (_Float16)tanhf(c[rt][r] + bv);
                else      Cf[(r0 + r) * ncol + nloc] = c[rt][r];
            }
        }
    } else {                             // only the last block takes this path
#pragma unroll
        for (int rt = 0; rt < 4; ++rt) {
            const int rloc = rt * 16 + mbase;
#pragma unroll
            for (int r = 0; r < 8; ++r) {
                if (rloc + r < valid) {
                    const int row = rowBase + rloc + r;
                    if (FUSE) Ch[row * ncol + nloc] = (_Float16)tanhf(c[rt][r] + bv);
                    else      Cf[row * ncol + nloc] = c[rt][r];
                }
            }
        }
    }
}

// ---------------- GCN scatter-aggregate ----------------
// 2 edges per 256-thread block; 128 lanes cover features.
__global__ void k_agg(const int* __restrict__ src, const int* __restrict__ dst,
                      const float* __restrict__ dinv, const float* __restrict__ hw,
                      float* __restrict__ agg, int E) {
    int e = blockIdx.x * 2 + (threadIdx.x >> 7);
    if (e >= E) return;
    int f = threadIdx.x & 127;
    int s = src[e], d = dst[e];
    float nrm = dinv[s] * dinv[d];
    atomicAdd(&agg[d * 128 + f], hw[s * 128 + f] * nrm);
}

// act = tanh(agg + hw*dinv^2 + bias)
__global__ void k_combine(const float* __restrict__ agg, const float* __restrict__ hw,
                          const float* __restrict__ dinv, const float* __restrict__ bias,
                          _Float16* __restrict__ act, int N) {
    int idx = blockIdx.x * blockDim.x + threadIdx.x;
    if (idx >= N * 128) return;
    int i = idx >> 7, f = idx & 127;
    float sn = dinv[i] * dinv[i];
    act[idx] = (_Float16)tanhf(agg[idx] + hw[idx] * sn + bias[f]);
}

__global__ void k_counts(const int* __restrict__ batch, float* __restrict__ cnts, int N) {
    int i = blockIdx.x * blockDim.x + threadIdx.x;
    int st = gridDim.x * blockDim.x;
    for (; i < N; i += st) atomicAdd(&cnts[batch[i]], 1.0f);
}

// final 32->1 linear fused with segment-sum pooling
__global__ void k_f3pool(const _Float16* __restrict__ act32, const float* __restrict__ Wf3,
                         const float* __restrict__ bf3, const int* __restrict__ batch,
                         float* __restrict__ usum, int N) {
    int i = blockIdx.x * blockDim.x + threadIdx.x;
    int st = gridDim.x * blockDim.x;
    for (; i < N; i += st) {
        float acc = bf3[0];
#pragma unroll
        for (int c = 0; c < 32; ++c) acc += (float)act32[i * 32 + c] * Wf3[c];
        atomicAdd(&usum[batch[i]], acc);
    }
}

__global__ void k_final(const float* __restrict__ usum, const float* __restrict__ cnts,
                        float* __restrict__ util, int G) {
    int g = blockIdx.x * blockDim.x + threadIdx.x;
    if (g < G) util[g] = usum[g] / fmaxf(cnts[g], 1.0f);
}

__global__ void k_pairs(const float* __restrict__ util, const int* __restrict__ ia,
                        const int* __restrict__ ib, float* __restrict__ out, int P) {
    int p = blockIdx.x * blockDim.x + threadIdx.x;
    if (p < P) {
        float d = util[ib[p]] - util[ia[p]];
        out[p] = 1.0f / (1.0f + expf(-d));
    }
}

// ---------------- launch ----------------

extern "C" void kernel_launch(void* const* d_in, const int* in_sizes, int n_in,
                              void* d_out, int out_size, void* d_ws, size_t ws_size,
                              hipStream_t stream) {
    const float* x     = (const float*)d_in[0];
    const int*   ei    = (const int*)d_in[1];
    const int*   batch = (const int*)d_in[2];
    const int*   idx_a = (const int*)d_in[3];
    const int*   idx_b = (const int*)d_in[4];
    const float* W_in  = (const float*)d_in[5];
    const float* b_in  = (const float*)d_in[6];
    const float* W1    = (const float*)d_in[7];
    const float* b1    = (const float*)d_in[8];
    const float* W2    = (const float*)d_in[9];
    const float* b2    = (const float*)d_in[10];
    const float* Wf1   = (const float*)d_in[11];
    const float* bf1   = (const float*)d_in[12];
    const float* Wf2   = (const float*)d_in[13];
    const float* bf2   = (const float*)d_in[14];
    const float* Wf3   = (const float*)d_in[15];
    const float* bf3   = (const float*)d_in[16];

    const int N = in_sizes[0] / FIN;
    const int E = in_sizes[1] / 2;
    const int P = in_sizes[3];
    const int G = out_size - P;
    const int* src = ei;
    const int* dst = ei + E;

    // workspace layout
    char* ws = (char*)d_ws;
    size_t off = 0;
    auto wsalloc = [&](size_t bytes) -> void* {
        void* p = ws + off;
        off += (bytes + 255) & ~(size_t)255;
        return p;
    };
    float*     dinv  = (float*)wsalloc((size_t)N * 4);            // deg -> dinv (in place)
    float*     hw    = (float*)wsalloc((size_t)N * 128 * 4);      // GEMM pre-activation
    float*     agg   = (float*)wsalloc((size_t)N * 128 * 4);      // scatter accumulator
    _Float16*  act   = (_Float16*)wsalloc((size_t)N * 128 * 2);   // f16 activations
    _Float16*  act32 = (_Float16*)wsalloc((size_t)N * 32 * 2);    // xpad / MLP-32 activations
    _Float16*  Winp  = (_Float16*)wsalloc(32 * 128 * 2);
    _Float16*  W1p   = (_Float16*)wsalloc(128 * 128 * 2);
    _Float16*  W2p   = (_Float16*)wsalloc(128 * 128 * 2);
    _Float16*  Wf1p  = (_Float16*)wsalloc(128 * 128 * 2);
    _Float16*  Wf2p  = (_Float16*)wsalloc(128 * 32 * 2);
    float*     usum  = (float*)wsalloc((size_t)G * 4);
    float*     cnts  = (float*)wsalloc((size_t)G * 4);

    float* out  = (float*)d_out;      // [0,P): pair sigmoids
    float* util = out + P;            // [P,P+G): per-graph utilities

    const int TB = 256;
    const int rowBlocks = (N + 63) / 64;

    // degree -> dinv
    k_zero<<<1024, TB, 0, stream>>>(dinv, N);
    k_deg<<<2048, TB, 0, stream>>>(dst, dinv, E);
    k_dinv<<<(N + TB - 1) / TB, TB, 0, stream>>>(dinv, N);

    // input padding + weight fragment packing
    k_padx<<<(N * 32 + TB - 1) / TB, TB, 0, stream>>>(x, act32, N);
    k_packB<<<(1 * 8 * 512 + TB - 1) / TB, TB, 0, stream>>>(W_in, Winp, 1, 8, 128, FIN);
    k_packB<<<(4 * 8 * 512 + TB - 1) / TB, TB, 0, stream>>>(W1,  W1p,  4, 8, 128, 128);
    k_packB<<<(4 * 8 * 512 + TB - 1) / TB, TB, 0, stream>>>(W2,  W2p,  4, 8, 128, 128);
    k_packB<<<(4 * 8 * 512 + TB - 1) / TB, TB, 0, stream>>>(Wf1, Wf1p, 4, 8, 128, 128);
    k_packB<<<(4 * 2 * 512 + TB - 1) / TB, TB, 0, stream>>>(Wf2, Wf2p, 4, 2, 32, 128);

    const int aggGrid = (E + 1) / 2;
    const int cmbGrid = (N * 128 + TB - 1) / TB;

    // GCN layer 1 (K=32 padded from F_IN=9)
    k_gemm_wmma<1, 8, false><<<rowBlocks, 256, 0, stream>>>(act32, Winp, hw, nullptr, nullptr, N);
    k_zero<<<4096, TB, 0, stream>>>(agg, (long)N * 128);
    k_agg<<<aggGrid, 256, 0, stream>>>(src, dst, dinv, hw, agg, E);
    k_combine<<<cmbGrid, TB, 0, stream>>>(agg, hw, dinv, b_in, act, N);

    // GCN layer 2
    k_gemm_wmma<4, 8, false><<<rowBlocks, 256, 0, stream>>>(act, W1p, hw, nullptr, nullptr, N);
    k_zero<<<4096, TB, 0, stream>>>(agg, (long)N * 128);
    k_agg<<<aggGrid, 256, 0, stream>>>(src, dst, dinv, hw, agg, E);
    k_combine<<<cmbGrid, TB, 0, stream>>>(agg, hw, dinv, b1, act, N);

    // GCN layer 3
    k_gemm_wmma<4, 8, false><<<rowBlocks, 256, 0, stream>>>(act, W2p, hw, nullptr, nullptr, N);
    k_zero<<<4096, TB, 0, stream>>>(agg, (long)N * 128);
    k_agg<<<aggGrid, 256, 0, stream>>>(src, dst, dinv, hw, agg, E);
    k_combine<<<cmbGrid, TB, 0, stream>>>(agg, hw, dinv, b2, act, N);

    // MLP head: 128->128 (fused tanh, in-place safe via LDS staging), 128->32 (fused tanh)
    k_gemm_wmma<4, 8, true><<<rowBlocks, 256, 0, stream>>>(act, Wf1p, nullptr, act, bf1, N);
    k_gemm_wmma<4, 2, true><<<rowBlocks, 64, 0, stream>>>(act, Wf2p, nullptr, act32, bf2, N);

    // 32->1 + mean pool + pairwise sigmoid
    k_zero<<<(G + TB - 1) / TB, TB, 0, stream>>>(usum, G);
    k_zero<<<(G + TB - 1) / TB, TB, 0, stream>>>(cnts, G);
    k_counts<<<1024, TB, 0, stream>>>(batch, cnts, N);
    k_f3pool<<<1024, TB, 0, stream>>>(act32, Wf3, bf3, batch, usum, N);
    k_final<<<(G + TB - 1) / TB, TB, 0, stream>>>(usum, cnts, util, G);
    k_pairs<<<(P + TB - 1) / TB, TB, 0, stream>>>(util, idx_a, idx_b, out, P);
}
